// SwiGLUSparseMoe_8624294331063
// MI455X (gfx1250) — matrix operand order repaired and checked by reading, not compiled, not run on hardware
//
#include <hip/hip_runtime.h>
#include <hip/hip_bf16.h>
#include <math.h>

// ---------------------------------------------------------------------------
// Sparse MoE SwiGLU (top-2 of 8) for MI455X / gfx1250.
//   T=8192, D=1024, H=2816, E=8, A=2.  ~283 GFLOP routed (vs 1.13 TF dense).
// bf16 v_wmma_f32_16x16x32_bf16 with fp32 accum; x + weights pre-converted to
// bf16 in workspace; A-tiles staged to LDS via global_load_async_to_lds_b128
// (double-buffered, s_wait_asynccnt); 64x128 output tiles for weight reuse;
// native global_atomic_add_f32 for the 2-way output combine.
// Workspace requirement: ~240 MB.
// ---------------------------------------------------------------------------

#define NEXP 8
#define NACT 2
#define DIM 1024
#define HID 2816
#define NTOK 8192                        // B*S
#define CAP (NTOK * NACT + NEXP * 64)    // 16896: segments padded to 64
#define CAP_TILES (CAP / 64)             // 264 row tiles of 64 slots
#define WELEM (NEXP * HID * DIM)         // elems per (E,H,D) weight tensor

// workspace byte offsets
#define OFF_COUNTS   0u
#define OFF_CURSORS  256u
#define OFF_POFFS    512u
#define OFF_ETILE    1024u
#define OFF_TLIST    8192u
#define OFF_PLIST    (OFF_TLIST + 67584u)
#define OFF_TOKE     (OFF_PLIST + 67584u)
#define OFF_TOKP     (OFF_TOKE + 65536u)
#define OFF_ZBUF     (OFF_TOKP + 65536u)
#define OFF_XB       (OFF_ZBUF + 256u)                 // NTOK*DIM bf16 (16 MB)
#define OFF_UPB      (OFF_XB + 16777216u)              // WELEM bf16 (44 MB)
#define OFF_GPB      (OFF_UPB + 46137344u)
#define OFF_DNB      (OFF_GPB + 46137344u)
#define OFF_HBUF     (OFF_DNB + 46137344u)             // CAP*HID bf16 (~91 MB)

typedef __attribute__((ext_vector_type(16))) __bf16 v16bf;
typedef __attribute__((ext_vector_type(8)))  __bf16 v8bf;
typedef __attribute__((ext_vector_type(4)))  __bf16 v4bf;
typedef __attribute__((ext_vector_type(8)))  float  v8f;

// B fragment: two contiguous 8-elem runs [kb..kb+7] and [kb+16..kb+23]
__device__ inline v16bf ldb16(const __bf16* p) {
  v8bf lo = *(const v8bf*)p;
  v8bf hi = *(const v8bf*)(p + 16);
  return __builtin_shufflevector(lo, hi, 0, 1, 2, 3, 4, 5, 6, 7, 8, 9, 10, 11,
                                 12, 13, 14, 15);
}
// A fragment: two contiguous 8-elem runs [kb..kb+7] and [kb+16..kb+23]
__device__ inline v16bf lda16(const __bf16* p) {
  v8bf lo = *(const v8bf*)p;
  v8bf hi = *(const v8bf*)(p + 16);
  return __builtin_shufflevector(lo, hi, 0, 1, 2, 3, 4, 5, 6, 7, 8, 9, 10, 11,
                                 12, 13, 14, 15);
}
__device__ inline void async_lds_b128(unsigned lds_addr, unsigned long long ga) {
  asm volatile("global_load_async_to_lds_b128 %0, %1, off"
               :: "v"(lds_addr), "v"(ga) : "memory");
}
__device__ inline void wait_async0() {
  asm volatile("s_wait_asynccnt 0" ::: "memory");
}

// ---------------------------------------------------------------------------
// 0) init: zero output, counters, zero-buffer; invalidate token list
// ---------------------------------------------------------------------------
__global__ __launch_bounds__(256) void moe_init(float* __restrict__ y,
                                                int* __restrict__ tlist,
                                                int* __restrict__ counts,
                                                float* __restrict__ zbuf) {
  int idx = blockIdx.x * 256 + threadIdx.x;
  if (idx < NTOK * DIM) y[idx] = 0.0f;
  if (idx < CAP) tlist[idx] = -1;
  if (idx < NEXP) counts[idx] = 0;
  if (idx < 64) zbuf[idx] = 0.0f;
}

// ---------------------------------------------------------------------------
// 0b) fp32 -> bf16 bulk conversion (x and the three weight tensors)
// ---------------------------------------------------------------------------
__global__ __launch_bounds__(256) void conv_bf16(const float* __restrict__ src,
                                                 __bf16* __restrict__ dst, int n) {
  int i = (blockIdx.x * 256 + threadIdx.x) * 4;
  if (i >= n) return;
  float4 v = *(const float4*)(src + i);
  v4bf o = {(__bf16)v.x, (__bf16)v.y, (__bf16)v.z, (__bf16)v.w};
  *(v4bf*)(dst + i) = o;
}

// ---------------------------------------------------------------------------
// 1) router: one wave per token; lane = g*8 + e
// ---------------------------------------------------------------------------
__global__ __launch_bounds__(256) void moe_router(const float* __restrict__ x,
                                                  const float* __restrict__ gw,
                                                  const float* __restrict__ gb,
                                                  int* __restrict__ counts,
                                                  int* __restrict__ tok_e,
                                                  float* __restrict__ tok_p) {
  int wave = threadIdx.x >> 5, lane = threadIdx.x & 31;
  int t = blockIdx.x * 8 + wave;
  int e = lane & 7, g = lane >> 3;

  const float* xr = x + (size_t)t * DIM;
  const float* wr = gw + (size_t)e * DIM;
  float acc = 0.0f;
  for (int d = g; d < DIM; d += 4) acc = fmaf(xr[d], wr[d], acc);
  acc += __shfl_xor(acc, 8, 32);
  acc += __shfl_xor(acc, 16, 32);
  float logit = acc + gb[e];

  float l[NEXP];
#pragma unroll
  for (int i = 0; i < NEXP; ++i) l[i] = __shfl(logit, i, 32);

  if (lane == 0) {
    float mx = l[0];
#pragma unroll
    for (int i = 1; i < NEXP; ++i) mx = fmaxf(mx, l[i]);
    float p[NEXP], sum = 0.0f;
#pragma unroll
    for (int i = 0; i < NEXP; ++i) { p[i] = __expf(l[i] - mx); sum += p[i]; }
    float inv = 1.0f / sum;
#pragma unroll
    for (int i = 0; i < NEXP; ++i) p[i] *= inv;
    int e0 = 0;
#pragma unroll
    for (int i = 1; i < NEXP; ++i) if (p[i] > p[e0]) e0 = i;
    int e1 = (e0 == 0) ? 1 : 0;
#pragma unroll
    for (int i = 0; i < NEXP; ++i)
      if (i != e0 && p[i] > p[e1]) e1 = i;
    float r = 1.0f / (p[e0] + p[e1]);
    atomicAdd(&counts[e0], 1);
    atomicAdd(&counts[e1], 1);
    tok_e[2 * t + 0] = e0;  tok_p[2 * t + 0] = p[e0] * r;
    tok_e[2 * t + 1] = e1;  tok_p[2 * t + 1] = p[e1] * r;
  }
}

// ---------------------------------------------------------------------------
// 2) scan: 64-padded prefix sum + tile->expert map
// ---------------------------------------------------------------------------
__global__ void moe_scan(const int* __restrict__ counts,
                         int* __restrict__ poffs, int* __restrict__ cursors,
                         int* __restrict__ etile) {
  if (threadIdx.x != 0) return;
  int run = 0;
  for (int e = 0; e < NEXP; ++e) {
    poffs[e] = run;
    cursors[e] = run;
    run += (counts[e] + 63) & ~63;
  }
  poffs[NEXP] = run;
  for (int e = 0; e < NEXP; ++e)
    for (int tt = poffs[e] >> 6; tt < (poffs[e + 1] >> 6); ++tt) etile[tt] = e;
  for (int tt = run >> 6; tt < CAP_TILES; ++tt) etile[tt] = -1;
}

// ---------------------------------------------------------------------------
// 3) scatter tokens into per-expert segments
// ---------------------------------------------------------------------------
__global__ __launch_bounds__(256) void moe_scatter(const int* __restrict__ tok_e,
                                                   const float* __restrict__ tok_p,
                                                   int* __restrict__ cursors,
                                                   int* __restrict__ tlist,
                                                   float* __restrict__ plist) {
  int t = blockIdx.x * 256 + threadIdx.x;
  if (t >= NTOK) return;
#pragma unroll
  for (int a = 0; a < NACT; ++a) {
    int e = tok_e[2 * t + a];
    int pos = atomicAdd(&cursors[e], 1);
    tlist[pos] = t;
    plist[pos] = tok_p[2 * t + a];
  }
}

// ---------------------------------------------------------------------------
// 4) mlp1: h = silu(x Wg^T) * (x Wu^T)   bf16 in -> bf16 out
//    block = 256 thr (8 waves); tile 64 slots x 128 hidden; K over D=1024 in
//    128-wide LDS chunks, async-staged + double-buffered.
// ---------------------------------------------------------------------------
__global__ __launch_bounds__(256) void moe_mlp1(const __bf16* __restrict__ xb,
                                                const __bf16* __restrict__ upb,
                                                const __bf16* __restrict__ gpb,
                                                const int* __restrict__ etile,
                                                const int* __restrict__ tlist,
                                                const float* __restrict__ zbuf,
                                                __bf16* __restrict__ hbuf) {
  int bt = blockIdx.x;
  int e = etile[bt];
  if (e < 0) return;
  int hblk = blockIdx.y * 128;
  int tid = threadIdx.x, wave = tid >> 5, lane = tid & 31;

  __shared__ __bf16 sxb[2][64][136];  // 272B row stride: 68 dw -> conflict-free
  __shared__ int stok[64];
  if (tid < 64) stok[tid] = tlist[bt * 64 + tid];
  __syncthreads();

  const int srow = tid >> 4;            // 0..15  (staging row within pass)
  const int scb = (tid & 15) * 16;      // byte column within 256B row
  unsigned long long zba = (unsigned long long)(size_t)zbuf + scb;

  // stage chunk (k0) into buffer b: 64 rows x 128 bf16 (gathered token rows)
  auto stage = [&](int k0, int b) {
#pragma unroll
    for (int p = 0; p < 4; ++p) {
      int row = p * 16 + srow;
      int tok = stok[row];
      unsigned long long ga =
          (tok >= 0)
              ? (unsigned long long)(size_t)(xb + (size_t)tok * DIM + k0) + scb
              : zba;
      unsigned la = (unsigned)(size_t)&sxb[b][row][0] + scb;
      async_lds_b128(la, ga);
    }
  };

  const int m = lane & 15;
  const int kb = (lane >> 4) << 3;       // A K-base: 0 / 8
  const int kbB = (lane >> 4) << 3;      // B K-base: 0 / 8
  const int hrow = hblk + wave * 16 + m; // B row (n = lane&15)
  const __bf16* wg = gpb + (size_t)e * HID * DIM + (size_t)hrow * DIM;
  const __bf16* wu = upb + (size_t)e * HID * DIM + (size_t)hrow * DIM;

  v8f accG[4] = {{0}, {0}, {0}, {0}};
  v8f accU[4] = {{0}, {0}, {0}, {0}};

  stage(0, 0);
  for (int c = 0; c < DIM / 128; ++c) {
    wait_async0();
    __syncthreads();
    if (c + 1 < DIM / 128) stage((c + 1) * 128, (c + 1) & 1);
    int k0 = c * 128, b = c & 1;
    __builtin_prefetch(wg + k0 + 128, 0, 3);
    __builtin_prefetch(wu + k0 + 128, 0, 3);
#pragma unroll
    for (int kk = 0; kk < 128; kk += 32) {
      v16bf bg = ldb16(wg + k0 + kk + kbB);
      v16bf bu = ldb16(wu + k0 + kk + kbB);
#pragma unroll
      for (int mt = 0; mt < 4; ++mt) {
        v16bf af = lda16(&sxb[b][mt * 16 + m][kk + kb]);
        accG[mt] = __builtin_amdgcn_wmma_f32_16x16x32_bf16(
            false, af, false, bg, (short)0, accG[mt], false, false);
        accU[mt] = __builtin_amdgcn_wmma_f32_16x16x32_bf16(
            false, af, false, bu, (short)0, accU[mt], false, false);
      }
    }
    __syncthreads();
  }

  int nout = hblk + wave * 16 + (lane & 15);
  int mbase = (lane < 16) ? 0 : 8;
#pragma unroll
  for (int mt = 0; mt < 4; ++mt)
#pragma unroll
    for (int i = 0; i < 8; ++i) {
      float g = accG[mt][i], u = accU[mt][i];
      float hv = (g / (1.0f + __expf(-g))) * u;  // silu(g)*u
      int slot = bt * 64 + mt * 16 + mbase + i;
      hbuf[(size_t)slot * HID + nout] = (__bf16)hv;
    }
}

// ---------------------------------------------------------------------------
// 5) mlp2: y[tok,:] += prob * (h @ Wd^T); tile 64 slots x 128 dim; K over H.
//    Native f32 fadd atomics: exactly two commutative contributions/element.
// ---------------------------------------------------------------------------
__global__ __launch_bounds__(256) void moe_mlp2(const __bf16* __restrict__ hbuf,
                                                const __bf16* __restrict__ dnb,
                                                const int* __restrict__ etile,
                                                const int* __restrict__ tlist,
                                                const float* __restrict__ plist,
                                                float* __restrict__ y) {
  int bt = blockIdx.x;
  int e = etile[bt];
  if (e < 0) return;
  int dblk = blockIdx.y * 128;
  int tid = threadIdx.x, wave = tid >> 5, lane = tid & 31;

  __shared__ __bf16 shb[2][64][136];
  __shared__ int stok[64];
  __shared__ float sp[64];
  if (tid < 64) {
    stok[tid] = tlist[bt * 64 + tid];
    sp[tid] = plist[bt * 64 + tid];
  }
  __syncthreads();

  const int srow = tid >> 4;
  const int scb = (tid & 15) * 16;
  auto stage = [&](int k0, int b) {
#pragma unroll
    for (int p = 0; p < 4; ++p) {
      int row = p * 16 + srow;  // padded slots hold zeros written by mlp1
      unsigned long long ga =
          (unsigned long long)(size_t)(hbuf + (size_t)(bt * 64 + row) * HID + k0) +
          scb;
      unsigned la = (unsigned)(size_t)&shb[b][row][0] + scb;
      async_lds_b128(la, ga);
    }
  };

  const int m = lane & 15;
  const int kb = (lane >> 4) << 3;
  const int kbB = (lane >> 4) << 3;
  const int drow = dblk + wave * 16 + m;
  const __bf16* wd = dnb + (size_t)e * DIM * HID + (size_t)drow * HID;

  v8f acc[4] = {{0}, {0}, {0}, {0}};

  stage(0, 0);
  for (int c = 0; c < HID / 128; ++c) {
    wait_async0();
    __syncthreads();
    if (c + 1 < HID / 128) stage((c + 1) * 128, (c + 1) & 1);
    int k0 = c * 128, b = c & 1;
    __builtin_prefetch(wd + k0 + 128, 0, 3);
#pragma unroll
    for (int kk = 0; kk < 128; kk += 32) {
      v16bf bf = ldb16(wd + k0 + kk + kbB);
#pragma unroll
      for (int mt = 0; mt < 4; ++mt) {
        v16bf af = lda16(&shb[b][mt * 16 + m][kk + kb]);
        acc[mt] = __builtin_amdgcn_wmma_f32_16x16x32_bf16(
            false, af, false, bf, (short)0, acc[mt], false, false);
      }
    }
    __syncthreads();
  }

  int ncol = dblk + wave * 16 + (lane & 15);
  int mbase = (lane < 16) ? 0 : 8;
#pragma unroll
  for (int mt = 0; mt < 4; ++mt)
#pragma unroll
    for (int i = 0; i < 8; ++i) {
      int r = mt * 16 + mbase + i;
      int tok = stok[r];
      if (tok >= 0)
        unsafeAtomicAdd(&y[(size_t)tok * DIM + ncol], sp[r] * acc[mt][i]);
    }
}

// ---------------------------------------------------------------------------
extern "C" void kernel_launch(void* const* d_in, const int* in_sizes, int n_in,
                              void* d_out, int out_size, void* d_ws, size_t ws_size,
                              hipStream_t stream) {
  const float* x   = (const float*)d_in[0];
  const float* gw  = (const float*)d_in[1];
  const float* gb  = (const float*)d_in[2];
  const float* upw = (const float*)d_in[3];
  const float* gpw = (const float*)d_in[4];
  const float* dnw = (const float*)d_in[5];
  float* y = (float*)d_out;

  char* ws = (char*)d_ws;
  int*    counts  = (int*)(ws + OFF_COUNTS);
  int*    cursors = (int*)(ws + OFF_CURSORS);
  int*    poffs   = (int*)(ws + OFF_POFFS);
  int*    etile   = (int*)(ws + OFF_ETILE);
  int*    tlist   = (int*)(ws + OFF_TLIST);
  float*  plist   = (float*)(ws + OFF_PLIST);
  int*    tok_e   = (int*)(ws + OFF_TOKE);
  float*  tok_p   = (float*)(ws + OFF_TOKP);
  float*  zbuf    = (float*)(ws + OFF_ZBUF);
  __bf16* xb      = (__bf16*)(ws + OFF_XB);
  __bf16* upb     = (__bf16*)(ws + OFF_UPB);
  __bf16* gpb     = (__bf16*)(ws + OFF_GPB);
  __bf16* dnb     = (__bf16*)(ws + OFF_DNB);
  __bf16* hbuf    = (__bf16*)(ws + OFF_HBUF);

  moe_init<<<dim3((NTOK * DIM + 255) / 256), 256, 0, stream>>>(y, tlist, counts, zbuf);
  conv_bf16<<<dim3(NTOK * DIM / 4 / 256), 256, 0, stream>>>(x, xb, NTOK * DIM);
  conv_bf16<<<dim3(WELEM / 4 / 256), 256, 0, stream>>>(upw, upb, WELEM);
  conv_bf16<<<dim3(WELEM / 4 / 256), 256, 0, stream>>>(gpw, gpb, WELEM);
  conv_bf16<<<dim3(WELEM / 4 / 256), 256, 0, stream>>>(dnw, dnb, WELEM);
  moe_router<<<dim3(NTOK / 8), 256, 0, stream>>>(x, gw, gb, counts, tok_e, tok_p);
  moe_scan<<<dim3(1), 32, 0, stream>>>(counts, poffs, cursors, etile);
  moe_scatter<<<dim3(NTOK / 256), 256, 0, stream>>>(tok_e, tok_p, cursors, tlist, plist);
  moe_mlp1<<<dim3(CAP_TILES, HID / 128), 256, 0, stream>>>(xb, upb, gpb, etile,
                                                           tlist, zbuf, hbuf);
  moe_mlp2<<<dim3(CAP_TILES, DIM / 128), 256, 0, stream>>>(hbuf, dnb, etile,
                                                           tlist, plist, y);
}
